// Model_28269474743137
// MI455X (gfx1250) — matrix operand order repaired and checked by reading, not compiled, not run on hardware
//
#include <hip/hip_runtime.h>
#include <stdint.h>

#define N_USERC  6040
#define N_MOVIEC 3883
#define NTOT     (N_USERC + N_MOVIEC)   // 9923
#define DIM      64
#define H1C      8
#define HC1      (H1C * DIM)            // 512
#define E_EDGES  500000
#define EL_EDGES 200000
#define MFEAT    18
#define NEG_SLOPE 0.2f
#define NTW      4                      // N-tiles per wave in GEMM

typedef __attribute__((ext_vector_type(16))) __bf16 v16bf;
typedef __attribute__((ext_vector_type(8)))  float  v8f;

union Frag16 { v16bf bf; uint32_t u[8]; uint4 q[2]; };

__device__ __forceinline__ uint16_t f2bf(float f) {
    uint32_t u = __float_as_uint(f);
    uint32_t r = u + 0x7FFFu + ((u >> 16) & 1u);   // round-to-nearest-even
    if ((u & 0x7F800000u) == 0x7F800000u) r = u;   // keep inf/nan payload
    return (uint16_t)(r >> 16);
}
// order-preserving float<->uint for atomicMax on floats
__device__ __forceinline__ uint32_t ford(float f) {
    uint32_t u = __float_as_uint(f);
    return (u >> 31) ? ~u : (u | 0x80000000u);
}
__device__ __forceinline__ float funord(uint32_t o) {
    uint32_t u = (o >> 31) ? (o ^ 0x80000000u) : ~o;
    return __uint_as_float(u);
}
__device__ __forceinline__ float lrelu(float x) { return x > 0.f ? x : NEG_SLOPE * x; }

// ---------------------------------------------------------------------------
// Node prep: X = concat(user_emb[gather], movie_x@Wlin + blin + movie_emb[gather])
// written directly as bf16 (feeds WMMA GEMM)
// ---------------------------------------------------------------------------
__global__ void k_prep(const float* __restrict__ movie_x, const int* __restrict__ unid,
                       const int* __restrict__ mnid, const float* __restrict__ user_emb,
                       const float* __restrict__ movie_emb, const float* __restrict__ Wlin,
                       const float* __restrict__ blin, uint16_t* __restrict__ Xbf) {
    int idx = blockIdx.x * blockDim.x + threadIdx.x;
    if (idx >= NTOT * DIM) return;
    int n = idx / DIM, d = idx % DIM;
    float v;
    if (n < N_USERC) {
        v = user_emb[unid[n] * DIM + d];
    } else {
        int m = n - N_USERC;
        float acc = blin[d];
        #pragma unroll
        for (int k = 0; k < MFEAT; ++k) acc += movie_x[m * MFEAT + k] * Wlin[k * DIM + d];
        v = acc + movie_emb[mnid[m] * DIM + d];
    }
    Xbf[idx] = f2bf(v);
}

// ---------------------------------------------------------------------------
// Pack weight W[K x N] (f32) into per-wave bf16 B-fragment layout:
// Ppack[((ntile*ksteps + kstep)*32 + lane)*8 + v] = {B[k,col], B[k+1,col]}
// with col = ntile*16 + lane%16, k = kstep*32 + (lane/16)*16 + 2v
// ---------------------------------------------------------------------------
__global__ void k_pack_w(const float* __restrict__ W, uint32_t* __restrict__ P, int K, int N) {
    int idx = blockIdx.x * blockDim.x + threadIdx.x;
    if (idx >= (K >> 1) * N) return;
    int ksteps = K >> 5;
    int v    = idx & 7;
    int lane = (idx >> 3) & 31;
    int t    = idx >> 8;              // ntile*ksteps + kstep
    int ks   = t % ksteps;
    int nt   = t / ksteps;
    int col  = nt * 16 + (lane & 15);
    int k    = ks * 32 + (lane >> 4) * 16 + 2 * v;
    uint32_t lo = f2bf(W[(size_t)k * N + col]);
    uint32_t hi = f2bf(W[(size_t)(k + 1) * N + col]);
    P[idx] = lo | (hi << 16);
}

// ---------------------------------------------------------------------------
// GEMM C[MxN] = A[MxK] * B[KxN], bf16 in / f32 accum.
// One wave computes a 16 x 64 strip (4 consecutive 16x16 N-tiles) so the
// A-fragment is reused by 4 independent v_wmma_f32_16x16x32_bf16 per k-step.
// All 4 B-fragments are loaded into distinct registers BEFORE the multiply
// chain so the loads issue as one clause and don't force a loadcnt drain
// before every WMMA.  Rows are clamped (not predicated) so the inner loop
// has no EXEC churn; out-of-range rows are filtered only at the store.
// ---------------------------------------------------------------------------
__global__ void __launch_bounds__(256) k_gemm_bf16(const uint32_t* __restrict__ Apairs,
                                                   const uint32_t* __restrict__ Bpack,
                                                   float* __restrict__ C, int M, int K, int N) {
    int lane = threadIdx.x & 31;
    int wid  = (blockIdx.x * blockDim.x + threadIdx.x) >> 5;
    int ngroups = (N >> 4) / NTW;
    int mtiles  = (M + 15) >> 4;
    if (wid >= mtiles * ngroups) return;          // wave-uniform: EXEC stays full
    int mt  = wid / ngroups;
    int nt0 = (wid - mt * ngroups) * NTW;
    int half = lane >> 4;
    int row  = mt * 16 + (lane & 15);
    int lrow = row < M ? row : M - 1;             // clamp: loads always valid
    int ksteps = K >> 5;
    int Kp = K >> 1;                              // uints per A row
    v8f c[NTW];
    #pragma unroll
    for (int j = 0; j < NTW; ++j) c[j] = (v8f){0.f,0.f,0.f,0.f,0.f,0.f,0.f,0.f};

    const uint32_t* ap    = Apairs + (size_t)lrow * Kp + half * 8;
    const uint32_t* bbase = Bpack + (size_t)nt0 * ksteps * 256 + (size_t)lane * 8;
    const size_t bstride  = (size_t)ksteps * 256; // uints between adjacent N-tiles

    for (int ks = 0; ks < ksteps; ++ks) {
        Frag16 a;
        a.q[0] = *(const uint4*)(ap);
        a.q[1] = *(const uint4*)(ap + 4);
        ap += 16;
        Frag16 b[NTW];
        #pragma unroll
        for (int j = 0; j < NTW; ++j) {
            const uint32_t* bp = bbase + (size_t)j * bstride;
            b[j].q[0] = *(const uint4*)(bp);
            b[j].q[1] = *(const uint4*)(bp + 4);
        }
        #pragma unroll
        for (int j = 0; j < NTW; ++j) {
            c[j] = __builtin_amdgcn_wmma_f32_16x16x32_bf16(false, a.bf, false, b[j].bf,
                                                           (short)0, c[j], false, false);
        }
        bbase += 256;
    }
    #pragma unroll
    for (int j = 0; j < NTW; ++j) {
        int gc = (nt0 + j) * 16 + (lane & 15);
        #pragma unroll
        for (int r = 0; r < 8; ++r) {
            int gr = mt * 16 + half * 8 + r;
            if (gr < M) C[(size_t)gr * N + gc] = c[j][r];
        }
    }
}

// ---------------------------------------------------------------------------
// Attention logits: out[n][type*H + h] = dot(Hsel[n, h*64 : h*64+64], a_sel[h])
// types: 0 = a_src(rates) on Hr, 1 = a_dst(rates) on Hr, 2 = a_src(rev) on Hv,
//        3 = a_dst(rev) on Hv.   64 threads / node.
// ---------------------------------------------------------------------------
__global__ void k_logits(const float* __restrict__ Hr, const float* __restrict__ Hv,
                         const float* __restrict__ as_r, const float* __restrict__ ad_r,
                         const float* __restrict__ as_v, const float* __restrict__ ad_v,
                         float* __restrict__ out, int H) {
    int n = blockIdx.x, t = threadIdx.x;
    int HC = H * 64;
    __shared__ float red[64];
    const float* Hsel[4] = {Hr, Hr, Hv, Hv};
    const float* asel[4] = {as_r, ad_r, as_v, ad_v};
    for (int type = 0; type < 4; ++type) {
        for (int h = 0; h < H; ++h) {
            red[t] = Hsel[type][(size_t)n * HC + h * 64 + t] * asel[type][h * 64 + t];
            __syncthreads();
            for (int s = 32; s > 0; s >>= 1) { if (t < s) red[t] += red[t + s]; __syncthreads(); }
            if (t == 0) out[(size_t)n * 4 * H + type * H + h] = red[0];
            __syncthreads();
        }
    }
}

// ---------------------------------------------------------------------------
// Edge pass 1: e = leaky(al_s[src]+al_d[dst]); store e; atomic segment-max.
// Handles both directions (rates: user->movie, rev: movie->user) per edge.
// ---------------------------------------------------------------------------
__global__ void k_edge_max(const int* __restrict__ esrc, const int* __restrict__ edst,
                           const float* __restrict__ L, float* __restrict__ e_r,
                           float* __restrict__ e_v, uint32_t* __restrict__ maxm,
                           uint32_t* __restrict__ maxu, int H) {
    int i = blockIdx.x * blockDim.x + threadIdx.x;
    if (i >= E_EDGES) return;
    int u = esrc[i], m = edst[i];
    int un = u, mn = N_USERC + m;
    for (int h = 0; h < H; ++h) {
        float er = lrelu(L[(size_t)un * 4 * H + 0 * H + h] + L[(size_t)mn * 4 * H + 1 * H + h]);
        e_r[(size_t)i * H + h] = er;
        atomicMax(&maxm[m * H + h], ford(er));
        float ev = lrelu(L[(size_t)mn * 4 * H + 2 * H + h] + L[(size_t)un * 4 * H + 3 * H + h]);
        e_v[(size_t)i * H + h] = ev;
        atomicMax(&maxu[u * H + h], ford(ev));
    }
}

// Edge pass 2: ex = exp(e - max[dst]); store ex in place; atomic segment-sum.
__global__ void k_edge_sum(const int* __restrict__ esrc, const int* __restrict__ edst,
                           float* __restrict__ e_r, float* __restrict__ e_v,
                           const uint32_t* __restrict__ maxm, const uint32_t* __restrict__ maxu,
                           float* __restrict__ denm, float* __restrict__ denu, int H) {
    int i = blockIdx.x * blockDim.x + threadIdx.x;
    if (i >= E_EDGES) return;
    int u = esrc[i], m = edst[i];
    for (int h = 0; h < H; ++h) {
        float ex = __expf(e_r[(size_t)i * H + h] - funord(maxm[m * H + h]));
        e_r[(size_t)i * H + h] = ex;
        atomicAdd(&denm[m * H + h], ex);
        float xv = __expf(e_v[(size_t)i * H + h] - funord(maxu[u * H + h]));
        e_v[(size_t)i * H + h] = xv;
        atomicAdd(&denu[u * H + h], xv);
    }
}

__global__ void k_recip(float* __restrict__ x, int n) {
    int i = blockIdx.x * blockDim.x + threadIdx.x;
    if (i < n) { float v = x[i]; x[i] = v > 0.f ? 1.0f / v : 0.f; }
}

// ---------------------------------------------------------------------------
// Edge pass 3: acc[dst] += alpha * hs[src].  One wave / edge; each 32-lane
// iteration covers half a head-block so alpha is wave-uniform.
// ---------------------------------------------------------------------------
__global__ void __launch_bounds__(256) k_edge_scatter(
        const int* __restrict__ esrc, const int* __restrict__ edst,
        const float* __restrict__ ex_r, const float* __restrict__ ex_v,
        const float* __restrict__ invm, const float* __restrict__ invu,
        const float* __restrict__ Hr, const float* __restrict__ Hv,
        float* __restrict__ accM, float* __restrict__ accU, int H) {
    int w = (blockIdx.x * blockDim.x + threadIdx.x) >> 5;
    int lane = threadIdx.x & 31;
    if (w >= E_EDGES) return;
    int u = esrc[w], m = edst[w];
    int HC = H * 64;
    const float* hsr = Hr + (size_t)u * HC;               // rates: src = user row
    const float* hsv = Hv + (size_t)(N_USERC + m) * HC;   // rev:   src = movie row
    __builtin_prefetch(hsr + lane, 0, 0);                 // global_prefetch_b8
    __builtin_prefetch(hsv + lane, 0, 0);
    float* outr = accM + (size_t)m * HC;
    float* outv = accU + (size_t)u * HC;
    for (int it = 0; it < (HC >> 5); ++it) {
        int h = it >> 1;
        int off = it * 32 + lane;
        float ar = ex_r[(size_t)w * H + h] * invm[m * H + h];
        float av = ex_v[(size_t)w * H + h] * invu[u * H + h];
        atomicAdd(outr + off, ar * hsr[off]);
        atomicAdd(outv + off, av * hsv[off]);
    }
}

// Epilogue layer 1: X2 = relu(acc + bias), concat(user, movie), written as bf16.
__global__ void k_epi1(const float* __restrict__ accU, const float* __restrict__ accM,
                       const float* __restrict__ b_rev, const float* __restrict__ b_rates,
                       uint16_t* __restrict__ X2bf) {
    int idx = blockIdx.x * blockDim.x + threadIdx.x;
    if (idx >= NTOT * HC1) return;
    int n = idx / HC1, j = idx % HC1;
    float v = (n < N_USERC)
                ? fmaxf(accU[(size_t)n * HC1 + j] + b_rev[j], 0.f)
                : fmaxf(accM[(size_t)(n - N_USERC) * HC1 + j] + b_rates[j], 0.f);
    X2bf[idx] = f2bf(v);
}

__global__ void k_bias(float* __restrict__ acc, const float* __restrict__ b, int total, int C) {
    int idx = blockIdx.x * blockDim.x + threadIdx.x;
    if (idx < total) acc[idx] += b[idx % C];
}

// Classifier: out[i] = dot(u2[label_src[i]], m2[label_dst[i]]) — wave per edge.
__global__ void __launch_bounds__(256) k_cls(const int* __restrict__ ls, const int* __restrict__ ld,
                                             const float* __restrict__ u2, const float* __restrict__ m2,
                                             float* __restrict__ out) {
    int w = (blockIdx.x * blockDim.x + threadIdx.x) >> 5;
    int lane = threadIdx.x & 31;
    if (w >= EL_EDGES) return;
    const float* a = u2 + (size_t)ls[w] * DIM;
    const float* b = m2 + (size_t)ld[w] * DIM;
    float s = a[lane] * b[lane] + a[lane + 32] * b[lane + 32];
    for (int o = 16; o > 0; o >>= 1) s += __shfl_down(s, o, 32);
    if (lane == 0) out[w] = s;
}

// ---------------------------------------------------------------------------
extern "C" void kernel_launch(void* const* d_in, const int* in_sizes, int n_in,
                              void* d_out, int out_size, void* d_ws, size_t ws_size,
                              hipStream_t stream) {
    const float* movie_x   = (const float*)d_in[0];
    const int*   unid      = (const int*)d_in[1];
    const int*   mnid      = (const int*)d_in[2];
    const int*   esrc      = (const int*)d_in[3];
    const int*   edst      = (const int*)d_in[4];
    const int*   lsrc      = (const int*)d_in[5];
    const int*   ldst      = (const int*)d_in[6];
    const float* user_emb  = (const float*)d_in[7];
    const float* movie_emb = (const float*)d_in[8];
    const float* Wlin      = (const float*)d_in[9];
    const float* blin      = (const float*)d_in[10];
    const float* W1_rates  = (const float*)d_in[11];
    const float* a1s_rates = (const float*)d_in[12];
    const float* a1d_rates = (const float*)d_in[13];
    const float* b1_rates  = (const float*)d_in[14];
    const float* W1_rev    = (const float*)d_in[15];
    const float* a1s_rev   = (const float*)d_in[16];
    const float* a1d_rev   = (const float*)d_in[17];
    const float* b1_rev    = (const float*)d_in[18];
    const float* W2_rates  = (const float*)d_in[19];
    const float* a2s_rates = (const float*)d_in[20];
    const float* a2d_rates = (const float*)d_in[21];
    const float* b2_rates  = (const float*)d_in[22];
    const float* W2_rev    = (const float*)d_in[23];
    const float* a2s_rev   = (const float*)d_in[24];
    const float* a2d_rev   = (const float*)d_in[25];
    const float* b2_rev    = (const float*)d_in[26];
    float* out = (float*)d_out;

    // ---- workspace bump allocator (256B aligned) ----
    size_t off = 0;
    auto alloc = [&](size_t bytes) -> char* {
        char* p = (char*)d_ws + off;
        off += (bytes + 255) & ~(size_t)255;
        return p;
    };
    uint16_t* Xbf    = (uint16_t*)alloc((size_t)NTOT * DIM * 2);
    uint32_t* W1r_p  = (uint32_t*)alloc((size_t)DIM * HC1 * 2);
    uint32_t* W1v_p  = (uint32_t*)alloc((size_t)DIM * HC1 * 2);
    uint32_t* W2r_p  = (uint32_t*)alloc((size_t)HC1 * DIM * 2);
    uint32_t* W2v_p  = (uint32_t*)alloc((size_t)HC1 * DIM * 2);
    float*    H1r    = (float*)alloc((size_t)NTOT * HC1 * 4);
    float*    H1v    = (float*)alloc((size_t)NTOT * HC1 * 4);
    float*    L1     = (float*)alloc((size_t)NTOT * 4 * H1C * 4);
    uint32_t* maxm   = (uint32_t*)alloc((size_t)N_MOVIEC * H1C * 4);
    uint32_t* maxu   = (uint32_t*)alloc((size_t)N_USERC * H1C * 4);
    float*    denm   = (float*)alloc((size_t)N_MOVIEC * H1C * 4);
    float*    denu   = (float*)alloc((size_t)N_USERC * H1C * 4);
    float*    accM   = (float*)alloc((size_t)N_MOVIEC * HC1 * 4);
    float*    accU   = (float*)alloc((size_t)N_USERC * HC1 * 4);
    // per-edge buffers for layer 1 (32 MB) — layer-2 buffers overlay this region
    char*     eRegion = alloc((size_t)E_EDGES * H1C * 4 * 2);
    float*    e_r = (float*)eRegion;
    float*    e_v = e_r + (size_t)E_EDGES * H1C;
    // ---- layer-2 overlay inside eRegion (valid after layer-1 scatter) ----
    size_t o2 = 0;
    auto alloc2 = [&](size_t bytes) -> char* {
        char* p = eRegion + o2;
        o2 += (bytes + 255) & ~(size_t)255;
        return p;
    };
    uint16_t* X2bf  = (uint16_t*)alloc2((size_t)NTOT * HC1 * 2);
    float*    H2r   = (float*)alloc2((size_t)NTOT * DIM * 4);
    float*    H2v   = (float*)alloc2((size_t)NTOT * DIM * 4);
    float*    L2    = (float*)alloc2((size_t)NTOT * 4 * 4);
    float*    e2_r  = (float*)alloc2((size_t)E_EDGES * 4);
    float*    e2_v  = (float*)alloc2((size_t)E_EDGES * 4);
    uint32_t* max2m = (uint32_t*)alloc2((size_t)N_MOVIEC * 4);
    uint32_t* max2u = (uint32_t*)alloc2((size_t)N_USERC * 4);
    float*    den2m = (float*)alloc2((size_t)N_MOVIEC * 4);
    float*    den2u = (float*)alloc2((size_t)N_USERC * 4);
    float*    accM2 = (float*)alloc2((size_t)N_MOVIEC * DIM * 4);
    float*    accU2 = (float*)alloc2((size_t)N_USERC * DIM * 4);

    const int B = 256;
    // ---- layer-1 init (ordered-uint encoding of any real value is > 0) ----
    hipMemsetAsync(maxm, 0, (size_t)N_MOVIEC * H1C * 4, stream);
    hipMemsetAsync(maxu, 0, (size_t)N_USERC * H1C * 4, stream);
    hipMemsetAsync(denm, 0, (size_t)N_MOVIEC * H1C * 4, stream);
    hipMemsetAsync(denu, 0, (size_t)N_USERC * H1C * 4, stream);
    hipMemsetAsync(accM, 0, (size_t)N_MOVIEC * HC1 * 4, stream);
    hipMemsetAsync(accU, 0, (size_t)N_USERC * HC1 * 4, stream);

    // ---- node features + weight packing ----
    k_prep<<<(NTOT * DIM + B - 1) / B, B, 0, stream>>>(movie_x, unid, mnid, user_emb,
                                                       movie_emb, Wlin, blin, Xbf);
    int packN1 = DIM * HC1 / 2;
    k_pack_w<<<(packN1 + B - 1) / B, B, 0, stream>>>(W1_rates, W1r_p, DIM, HC1);
    k_pack_w<<<(packN1 + B - 1) / B, B, 0, stream>>>(W1_rev,   W1v_p, DIM, HC1);
    int packN2 = HC1 * DIM / 2;
    k_pack_w<<<(packN2 + B - 1) / B, B, 0, stream>>>(W2_rates, W2r_p, HC1, DIM);
    k_pack_w<<<(packN2 + B - 1) / B, B, 0, stream>>>(W2_rev,   W2v_p, HC1, DIM);

    // ---- layer-1 GEMMs: [9923,64] x [64,512] via v_wmma_f32_16x16x32_bf16 ----
    int mt1 = (NTOT + 15) / 16;
    int g1 = (mt1 * (HC1 / 16 / NTW) * 32 + B - 1) / B;
    k_gemm_bf16<<<g1, B, 0, stream>>>((const uint32_t*)Xbf, W1r_p, H1r, NTOT, DIM, HC1);
    k_gemm_bf16<<<g1, B, 0, stream>>>((const uint32_t*)Xbf, W1v_p, H1v, NTOT, DIM, HC1);

    // ---- layer-1 attention ----
    k_logits<<<NTOT, 64, 0, stream>>>(H1r, H1v, a1s_rates, a1d_rates, a1s_rev, a1d_rev, L1, H1C);
    int ge = (E_EDGES + B - 1) / B;
    k_edge_max<<<ge, B, 0, stream>>>(esrc, edst, L1, e_r, e_v, maxm, maxu, H1C);
    k_edge_sum<<<ge, B, 0, stream>>>(esrc, edst, e_r, e_v, maxm, maxu, denm, denu, H1C);
    k_recip<<<(N_MOVIEC * H1C + B - 1) / B, B, 0, stream>>>(denm, N_MOVIEC * H1C);
    k_recip<<<(N_USERC * H1C + B - 1) / B, B, 0, stream>>>(denu, N_USERC * H1C);
    k_edge_scatter<<<(E_EDGES * 32 + B - 1) / B, B, 0, stream>>>(esrc, edst, e_r, e_v,
        denm, denu, H1r, H1v, accM, accU, H1C);

    // ---- layer-1 epilogue: relu(acc + bias) -> bf16 X2 ----
    k_epi1<<<((size_t)NTOT * HC1 + B - 1) / B, B, 0, stream>>>(accU, accM, b1_rev, b1_rates, X2bf);

    // ---- layer-2 init (overlay region is dead now) ----
    hipMemsetAsync(max2m, 0, (size_t)N_MOVIEC * 4, stream);
    hipMemsetAsync(max2u, 0, (size_t)N_USERC * 4, stream);
    hipMemsetAsync(den2m, 0, (size_t)N_MOVIEC * 4, stream);
    hipMemsetAsync(den2u, 0, (size_t)N_USERC * 4, stream);
    hipMemsetAsync(accM2, 0, (size_t)N_MOVIEC * DIM * 4, stream);
    hipMemsetAsync(accU2, 0, (size_t)N_USERC * DIM * 4, stream);

    // ---- layer-2 GEMMs: [9923,512] x [512,64] ----
    int g2 = (mt1 * (DIM / 16 / NTW) * 32 + B - 1) / B;
    k_gemm_bf16<<<g2, B, 0, stream>>>((const uint32_t*)X2bf, W2r_p, H2r, NTOT, HC1, DIM);
    k_gemm_bf16<<<g2, B, 0, stream>>>((const uint32_t*)X2bf, W2v_p, H2v, NTOT, HC1, DIM);

    // ---- layer-2 attention (heads = 1) ----
    k_logits<<<NTOT, 64, 0, stream>>>(H2r, H2v, a2s_rates, a2d_rates, a2s_rev, a2d_rev, L2, 1);
    k_edge_max<<<ge, B, 0, stream>>>(esrc, edst, L2, e2_r, e2_v, max2m, max2u, 1);
    k_edge_sum<<<ge, B, 0, stream>>>(esrc, edst, e2_r, e2_v, max2m, max2u, den2m, den2u, 1);
    k_recip<<<(N_MOVIEC + B - 1) / B, B, 0, stream>>>(den2m, N_MOVIEC);
    k_recip<<<(N_USERC + B - 1) / B, B, 0, stream>>>(den2u, N_USERC);
    k_edge_scatter<<<(E_EDGES * 32 + B - 1) / B, B, 0, stream>>>(esrc, edst, e2_r, e2_v,
        den2m, den2u, H2r, H2v, accM2, accU2, 1);

    // ---- layer-2 bias (no relu), then classifier ----
    k_bias<<<(N_MOVIEC * DIM + B - 1) / B, B, 0, stream>>>(accM2, b2_rates, N_MOVIEC * DIM, DIM);
    k_bias<<<(N_USERC * DIM + B - 1) / B, B, 0, stream>>>(accU2, b2_rev, N_USERC * DIM, DIM);
    k_cls<<<(EL_EDGES * 32 + B - 1) / B, B, 0, stream>>>(lsrc, ldst, accU2, accM2, out);

    (void)in_sizes; (void)n_in; (void)out_size; (void)ws_size;
}
